// AqtDotGeneral_19481971655318
// MI455X (gfx1250) — 8-bit matrix operand order repaired and checked by reading, compiled, not run on hardware
//
#include <hip/hip_runtime.h>
#include <cstdint>
#include <cstddef>

typedef int v8i __attribute__((ext_vector_type(8)));

#define K_DIM 1024
#define N_DIM 1024
#define M_DIM (4 * 8192)

__device__ __forceinline__ int quant1(float x, float inv) {
    float q = rintf(x * inv);
    q = fminf(fmaxf(q, -127.0f), 127.0f);
    return (int)q;
}

__device__ __forceinline__ int pack4(float4 f, float inv) {
    int q0 = quant1(f.x, inv), q1 = quant1(f.y, inv);
    int q2 = quant1(f.z, inv), q3 = quant1(f.w, inv);
    return (q0 & 255) | ((q1 & 255) << 8) | ((q2 & 255) << 16) | ((q3 & 255) << 24);
}

__device__ __forceinline__ void async_load_b128(uint32_t lds_addr, uint64_t sbase,
                                                uint32_t voff) {
    asm volatile("global_load_async_to_lds_b128 %0, %1, %2"
                 :
                 : "v"(lds_addr), "v"(voff), "s"(sbase)
                 : "memory");
}

__device__ __forceinline__ void wait_async0() {
#if __has_builtin(__builtin_amdgcn_s_wait_asynccnt)
    __builtin_amdgcn_s_wait_asynccnt(0);
#else
    asm volatile("s_wait_asynccnt 0x0" ::: "memory");
#endif
}

__global__ __launch_bounds__(256) void quant_rhs_kernel(const float* __restrict__ rhs,
                                                        int8_t* __restrict__ qB,
                                                        float* __restrict__ bscale) {
    const int n = blockIdx.x * 256 + threadIdx.x;
    float amax = 0.0f;
    for (int k = 0; k < K_DIM; ++k)
        amax = fmaxf(amax, fabsf(rhs[(size_t)k * N_DIM + n]));
    const float s = fmaxf(amax, 1e-12f) * (1.0f / 127.0f);
    bscale[n] = s;
    const float inv = 1.0f / s;
    for (int k = 0; k < K_DIM; ++k) {
        float q = rintf(rhs[(size_t)k * N_DIM + n] * inv);
        q = fminf(fmaxf(q, -127.0f), 127.0f);
        qB[(size_t)n * K_DIM + k] = (int8_t)q;
    }
}

__global__ __launch_bounds__(256) void aqt_gemm_kernel(const float* __restrict__ lhs,
                                                       const int8_t* __restrict__ qB,
                                                       const float* __restrict__ bscale,
                                                       float* __restrict__ out) {
    __shared__ __align__(16) int8_t sB[2][16 * K_DIM];
    __shared__ float sScale[128];

    const int tid    = threadIdx.x;
    const int wave   = tid >> 5;
    const int lane   = tid & 31;
    const int half   = lane >> 4;
    const int lcol   = lane & 15;
    const int mLocal = wave * 16;
    const int mBase  = blockIdx.x * 128;

    const uint64_t qb_base = (uint64_t)(uintptr_t)qB;

#pragma unroll
    for (int i = 0; i < 4; ++i) {
        const uint32_t off = (uint32_t)(tid * 16 + i * 4096);
        async_load_b128((uint32_t)(uintptr_t)(sB[0] + off), qb_base, off);
    }

    for (int r = 0; r < 16; ++r) {
        const float* row = lhs + (size_t)(mBase + mLocal + r) * K_DIM;
        float amax = 0.0f;
#pragma unroll
        for (int it = 0; it < 8; ++it) {
            float4 v = *(const float4*)(row + lane * 4 + it * 128);
            amax = fmaxf(amax,
                         fmaxf(fmaxf(fabsf(v.x), fabsf(v.y)),
                               fmaxf(fabsf(v.z), fabsf(v.w))));
        }
#pragma unroll
        for (int off = 16; off > 0; off >>= 1)
            amax = fmaxf(amax, __shfl_xor(amax, off, 32));
        if (lane == 0)
            sScale[mLocal + r] = fmaxf(amax, 1e-12f) * (1.0f / 127.0f);
    }

    const float  invA = 1.0f / sScale[mLocal + lcol];
    const float* rowp = lhs + (size_t)(mBase + mLocal + lcol) * K_DIM;
    v8i afrag[16];
#pragma unroll
    for (int j = 0; j < 16; ++j) {
#pragma unroll
        for (int c = 0; c < 4; ++c) {
            const int off = j * 64 + (c >> 1) * 32 + (c & 1) * 8 + half * 16;
            float4 f0 = *(const float4*)(rowp + off);
            float4 f1 = *(const float4*)(rowp + off + 4);
            afrag[j][2 * c]     = pack4(f0, invA);
            afrag[j][2 * c + 1] = pack4(f1, invA);
        }
    }

    float aScale[8];
#pragma unroll
    for (int g = 0; g < 8; ++g)
        aScale[g] = sScale[mLocal + half * 8 + g];

    wait_async0();
    __syncthreads();

    for (int nt = 0; nt < 64; ++nt) {
        const int cur = nt & 1;

        if (nt + 1 < 64) {
            const int nxt = cur ^ 1;
#pragma unroll
            for (int i = 0; i < 4; ++i) {
                const uint32_t off = (uint32_t)(tid * 16 + i * 4096);
                async_load_b128((uint32_t)(uintptr_t)(sB[nxt] + off),
                                qb_base,
                                (uint32_t)((nt + 1) * 16384) + off);
            }
        }

        v8i acc = {0, 0, 0, 0, 0, 0, 0, 0};
        const int8_t* sBc = sB[cur];
#pragma unroll
        for (int j = 0; j < 16; ++j) {
            const int boff = lcol * K_DIM + j * 64 + half * 16;
            int4 b0 = *(const int4*)(sBc + boff);
            int4 b1 = *(const int4*)(sBc + boff + 32);
            v8i bf;
            bf[0] = b0.x; bf[1] = b0.y; bf[2] = b0.z; bf[3] = b0.w;
            bf[4] = b1.x; bf[5] = b1.y; bf[6] = b1.z; bf[7] = b1.w;
            acc = __builtin_amdgcn_wmma_i32_16x16x64_iu8(
                true, afrag[j], true, bf, acc, false, false);
        }

        const int   col  = nt * 16 + lcol;
        const float sOut = bscale[col];
#pragma unroll
        for (int g = 0; g < 8; ++g) {
            const int row = mBase + mLocal + half * 8 + g;
            out[(size_t)row * N_DIM + col] = (float)acc[g] * aScale[g] * sOut;
        }

        wait_async0();
        __syncthreads();
    }
}

extern "C" void kernel_launch(void* const* d_in, const int* in_sizes, int n_in,
                              void* d_out, int out_size, void* d_ws, size_t ws_size,
                              hipStream_t stream) {
    (void)in_sizes; (void)n_in; (void)out_size; (void)ws_size;
    const float* lhs = (const float*)d_in[0];
    const float* rhs = (const float*)d_in[1];
    float*       out = (float*)d_out;

    int8_t* qB     = (int8_t*)d_ws;
    float*  bscale = (float*)((char*)d_ws + (size_t)N_DIM * K_DIM);

    quant_rhs_kernel<<<N_DIM / 256, 256, 0, stream>>>(rhs, qB, bscale);
    aqt_gemm_kernel<<<M_DIM / 128, 256, 0, stream>>>(lhs, qB, bscale, out);
}
